// SpatialDecomposition_31645319037418
// MI455X (gfx1250) — compile-verified, matrix operand order checked by reading, not run on hardware
//
#include <hip/hip_runtime.h>
#include <hip/hip_bf16.h>

typedef __attribute__((ext_vector_type(16))) _Float16 v16h;
typedef __attribute__((ext_vector_type(8)))  float    v8f;
typedef _Float16 f16;

#define BT_TOT 128
#define NPATCH 256
#define DVIS   1024
#define NC     16      // branches*heads
#define SCALE  0.125f  // 64^-0.5

// ---------------- workspace layout (bytes) ----------------
#define WS_G16     0            // [1024][16] f16        32768
#define WS_S       32768        // [16] f32
#define WS_CQ      33024        // [16] f32
#define WS_LOSS    33280        // f32
#define WS_MU      33536        // [32768] f32           131072
#define WS_RS      164608       // [32768] f32           131072
#define WS_ATTN16  295680       // [128][16][256] f16    1048576
#define WS_BV2     1344256      // [1024][1024] f16      2097152  Bv2[c*64+j][d]
#define WS_WOT     3441408      // [4][256][256] f32     1048576  WoT[n][j2][k]

// =============== kernel 0: prepare folded weights ===============
__global__ __launch_bounds__(256) void k0_prep(
    const float* __restrict__ g, const float* __restrict__ be,
    const float* __restrict__ query, const float* __restrict__ Wk,
    const float* __restrict__ bk, const float* __restrict__ Wv,
    const float* __restrict__ Wo,
    f16* __restrict__ G16, float* __restrict__ S, float* __restrict__ Cq,
    f16* __restrict__ Bv2, float* __restrict__ WoT, float* __restrict__ lossAcc)
{
    int tid  = threadIdx.x;
    int gidx = blockIdx.x * 256 + tid;
    int nth  = gridDim.x * 256;

    // Bv2[cj][d] = Wv[n][d][rem]  (cj = n*256+rem, rem = h*64+j)
    for (int e = gidx; e < (1 << 20); e += nth) {
        int cj = e >> 10, d = e & 1023;
        int n = cj >> 8, rem = cj & 255;
        Bv2[e] = (f16)Wv[((size_t)n * 1024 + d) * 256 + rem];
    }
    // WoT[n][j2][k] = Wo[n][k][j2]
    for (int e = gidx; e < 4 * 256 * 256; e += nth) {
        int n = e >> 16, r = e & 65535, j2 = r >> 8, k = r & 255;
        WoT[e] = Wo[((size_t)n * 256 + k) * 256 + j2];
    }

    if (blockIdx.x == 0) {
        if (tid < 16) { S[tid] = 0.f; Cq[tid] = 0.f; }
        if (tid == 0) *lossAcc = 0.f;
        __syncthreads();
        // G16[d*16+c] = gamma[d] * (Wk[n][d][h*64:..] . query[c]); S,C via atomics
        for (int it = 0; it < 64; ++it) {
            int idx = tid + it * 256;
            int d = idx >> 4, c = idx & 15;
            int n = c >> 2, h = c & 3;
            const float* wkc = Wk + ((size_t)n * 1024 + d) * 256 + h * 64;
            const float* qv  = query + c * 64;
            float wq = 0.f;
            for (int j = 0; j < 64; ++j) wq += wkc[j] * qv[j];
            float Gv = g[d] * wq;
            G16[idx] = (f16)Gv;
            atomicAdd(&S[c], Gv);
            atomicAdd(&Cq[c], be[d] * wq);
        }
        if (tid < 16) {  // bias term bk . q  (order vs other atomics irrelevant)
            int c = tid, n = c >> 2, h = c & 3;
            const float* qv = query + c * 64;
            float bq = 0.f;
            for (int j = 0; j < 64; ++j) bq += bk[n * 256 + h * 64 + j] * qv[j];
            atomicAdd(&Cq[c], bq);
        }
    }
}

// =============== kernel 1: LN stats + scores WMMA + softmax + diversity ===============
__global__ __launch_bounds__(256) void k1_scores(
    const float* __restrict__ x, const f16* __restrict__ G16,
    const float* __restrict__ S, const float* __restrict__ Cq,
    float* __restrict__ mu_ws, float* __restrict__ rs_ws,
    f16* __restrict__ attn16, float* __restrict__ lossAcc)
{
    __shared__ f16   Gsm[DVIS * NC];        // 32 KB
    __shared__ float scoresSm[NC * NPATCH]; // 16 KB (scores -> attn in place)
    __shared__ float muSm[NPATCH], rsSm[NPATCH];
    __shared__ float aavgSm[4 * NPATCH];    // 4 KB
    __shared__ float normSm[4];

    int bt = blockIdx.x;
    int tid = threadIdx.x, lane = tid & 31, wave = tid >> 5;

    { // stage folded weights into LDS
        const uint32_t* src = (const uint32_t*)G16;
        uint32_t* dst = (uint32_t*)Gsm;
        for (int i = tid; i < DVIS * NC / 2; i += 256) dst[i] = src[i];
    }

    // ---- LayerNorm stats: one wave per patch, coalesced HBM pass over x ----
    for (int p = wave; p < NPATCH; p += 8) {
        const float* row = x + ((size_t)(bt * NPATCH + p)) * DVIS;
        float s = 0.f, s2 = 0.f;
        for (int i = lane; i < DVIS; i += 32) { float v = row[i]; s += v; s2 += v * v; }
        for (int off = 16; off > 0; off >>= 1) {
            s += __shfl_xor(s, off, 32); s2 += __shfl_xor(s2, off, 32);
        }
        float mu = s * (1.f / DVIS);
        float var = s2 * (1.f / DVIS) - mu * mu;
        float rs = rsqrtf(var + 1e-5f);
        if (lane == 0) {
            muSm[p] = mu; rsSm[p] = rs;
            mu_ws[bt * NPATCH + p] = mu; rs_ws[bt * NPATCH + p] = rs;
        }
    }
    __syncthreads();

    // ---- scores GEMM: x(256x1024, raw f32->f16) @ G(1024x16), f32 acc ----
    int cL = lane & 15;
    int kb = (lane >> 4) * 8;
    int mb = (lane >> 4) * 8;
    for (int mt = wave; mt < 16; mt += 8) {
        int prow = mt * 16 + (lane & 15);
        const float* arow = x + ((size_t)(bt * NPATCH + prow)) * DVIS;
        v8f acc = {};
        for (int kt = 0; kt < 32; ++kt) {
            int d0 = kt * 32 + kb;  // multiple of 8 -> 16B aligned
            const float4* a0 = (const float4*)(arow + d0);
            const float4* a1 = (const float4*)(arow + d0 + 16);
            float4 u0 = a0[0], u1 = a0[1], u2 = a1[0], u3 = a1[1];
            v16h a, b;
            a[0]=(f16)u0.x; a[1]=(f16)u0.y; a[2]=(f16)u0.z; a[3]=(f16)u0.w;
            a[4]=(f16)u1.x; a[5]=(f16)u1.y; a[6]=(f16)u1.z; a[7]=(f16)u1.w;
            a[8]=(f16)u2.x; a[9]=(f16)u2.y; a[10]=(f16)u2.z; a[11]=(f16)u2.w;
            a[12]=(f16)u3.x; a[13]=(f16)u3.y; a[14]=(f16)u3.z; a[15]=(f16)u3.w;
            int kk = d0;
            #pragma unroll
            for (int i = 0; i < 8; ++i) {
                b[i]     = Gsm[(kk + i) * NC + cL];
                b[8 + i] = Gsm[(kk + 16 + i) * NC + cL];
            }
            acc = __builtin_amdgcn_wmma_f32_16x16x32_f16(
                false, a, false, b, (short)0, acc, false, false);
        }
        float Sc = S[cL], Cc = Cq[cL];
        #pragma unroll
        for (int r = 0; r < 8; ++r) {
            int p = mt * 16 + mb + r;
            scoresSm[cL * NPATCH + p] =
                SCALE * (rsSm[p] * (acc[r] - muSm[p] * Sc) + Cc);
        }
    }
    __syncthreads();

    // ---- softmax over 256 patches, 2 rows per wave ----
    for (int c = wave; c < 16; c += 8) {
        float vals[8], m = -1e30f;
        #pragma unroll
        for (int j = 0; j < 8; ++j) {
            vals[j] = scoresSm[c * NPATCH + lane + j * 32];
            m = fmaxf(m, vals[j]);
        }
        for (int off = 16; off > 0; off >>= 1) m = fmaxf(m, __shfl_xor(m, off, 32));
        float s = 0.f;
        #pragma unroll
        for (int j = 0; j < 8; ++j) { vals[j] = __expf(vals[j] - m); s += vals[j]; }
        for (int off = 16; off > 0; off >>= 1) s += __shfl_xor(s, off, 32);
        float inv = 1.f / s;
        #pragma unroll
        for (int j = 0; j < 8; ++j) {
            float a = vals[j] * inv;
            scoresSm[c * NPATCH + lane + j * 32] = a;
            attn16[((size_t)bt * NC + c) * NPATCH + lane + j * 32] = (f16)a;
        }
    }
    __syncthreads();

    // ---- diversity loss: head-avg, normalize, pairwise dots ----
    for (int i = tid; i < 4 * NPATCH; i += 256) {
        int n = i >> 8, p = i & 255;
        aavgSm[i] = 0.25f * (scoresSm[(n * 4 + 0) * NPATCH + p] +
                             scoresSm[(n * 4 + 1) * NPATCH + p] +
                             scoresSm[(n * 4 + 2) * NPATCH + p] +
                             scoresSm[(n * 4 + 3) * NPATCH + p]);
    }
    __syncthreads();
    if (wave < 4) {
        float s = 0.f;
        #pragma unroll
        for (int j = 0; j < 8; ++j) {
            float v = aavgSm[wave * NPATCH + lane + j * 32]; s += v * v;
        }
        for (int off = 16; off > 0; off >>= 1) s += __shfl_xor(s, off, 32);
        if (lane == 0) normSm[wave] = fmaxf(sqrtf(s), 1e-8f);
    }
    __syncthreads();
    if (wave < 6) {
        const int pi[6] = {0, 0, 0, 1, 1, 2};
        const int pj[6] = {1, 2, 3, 2, 3, 3};
        int i = pi[wave], j = pj[wave];
        float s = 0.f;
        #pragma unroll
        for (int q = 0; q < 8; ++q)
            s += aavgSm[i * NPATCH + lane + q * 32] * aavgSm[j * NPATCH + lane + q * 32];
        for (int off = 16; off > 0; off >>= 1) s += __shfl_xor(s, off, 32);
        if (lane == 0) atomicAdd(lossAcc, s / (normSm[i] * normSm[j]));
    }
}

// =============== kernel 2: attn^T x_ln WMMA + Wv gather + Wo epilogue ===============
__global__ __launch_bounds__(256) void k2_pool(
    const float* __restrict__ x, const float* __restrict__ g,
    const float* __restrict__ be, const float* __restrict__ mu_ws,
    const float* __restrict__ rs_ws, const f16* __restrict__ attn16,
    const f16* __restrict__ Bv2, const float* __restrict__ WoT,
    const float* __restrict__ bv, const float* __restrict__ bo,
    float* __restrict__ out)
{
    __shared__ f16   attnSm[NC * NPATCH]; // 8 KB
    __shared__ f16   xnSm[NPATCH * 64];   // 32 KB
    __shared__ float ySm[NC * 64];        // 4 KB
    __shared__ float pooledSm[1024];      // 4 KB

    int bt = blockIdx.x;
    int tid = threadIdx.x, lane = tid & 31, wave = tid >> 5;

    { // stage this row's attention weights
        const uint32_t* src = (const uint32_t*)(attn16 + (size_t)bt * NC * NPATCH);
        uint32_t* dst = (uint32_t*)attnSm;
        for (int i = tid; i < NC * NPATCH / 2; i += 256) dst[i] = src[i];
    }

    float pa[4] = {0.f, 0.f, 0.f, 0.f};
    int cOwn = tid >> 4, j0 = (tid & 15) * 4;  // pooled ownership: (c, j0..j0+3)
    int dl = tid & 63, pr = tid >> 6;          // tile-load mapping
    int ntile = wave & 3, khalf = wave >> 2;   // WMMA work split
    int cA = lane & 15, kb = (lane >> 4) * 8, mbD = (lane >> 4) * 8;
    int nc = ntile * 16 + (lane & 15);

    for (int dt = 0; dt < 16; ++dt) {
        for (int i = tid; i < NC * 64; i += 256) ySm[i] = 0.f;
        // LayerNorm + f16 convert 256x64 tile (x is L2-resident after k1)
        int d = dt * 64 + dl;
        float gd = g[d], bd = be[d];
        for (int pp = 0; pp < 64; ++pp) {
            int p = pp * 4 + pr;
            float mu = mu_ws[bt * NPATCH + p], rs = rs_ws[bt * NPATCH + p];
            float v = x[((size_t)(bt * NPATCH + p)) * DVIS + d];
            xnSm[p * 64 + dl] = (f16)((v - mu) * rs * gd + bd);
        }
        __syncthreads();
        // Y(16x64) = attn(16x256) @ xn(256x64); 8 waves: 4 ntiles x 2 K-halves
        {
            v8f acc = {};
            for (int kt = khalf * 4; kt < khalf * 4 + 4; ++kt) {
                int kk = kt * 32 + kb;
                v16h a, b;
                #pragma unroll
                for (int i = 0; i < 8; ++i) {
                    a[i]     = attnSm[cA * NPATCH + kk + i];
                    a[8 + i] = attnSm[cA * NPATCH + kk + 16 + i];
                    b[i]     = xnSm[(kk + i) * 64 + nc];
                    b[8 + i] = xnSm[(kk + 16 + i) * 64 + nc];
                }
                acc = __builtin_amdgcn_wmma_f32_16x16x32_f16(
                    false, a, false, b, (short)0, acc, false, false);
            }
            #pragma unroll
            for (int r = 0; r < 8; ++r)
                atomicAdd(&ySm[(mbD + r) * 64 + nc], acc[r]);  // merge K-halves
        }
        __syncthreads();
        // pooled[c][j] += Y[c][:] . Wv-column  (Bv2 gathered, contiguous in d)
        {
            const f16* bvp = Bv2 + ((size_t)(cOwn * 64 + j0)) * DVIS + dt * 64;
            #pragma unroll
            for (int q = 0; q < 4; ++q) {
                const f16* bq = bvp + (size_t)q * DVIS;
                float s = 0.f;
                for (int d2 = 0; d2 < 64; ++d2)
                    s += ySm[cOwn * 64 + d2] * (float)bq[d2];
                pa[q] += s;
            }
        }
        __syncthreads();
    }

    { // + bv bias (sum(attn)==1)
        int n = cOwn >> 2, h = cOwn & 3;
        #pragma unroll
        for (int q = 0; q < 4; ++q)
            pooledSm[cOwn * 64 + j0 + q] = pa[q] + bv[n * 256 + h * 64 + j0 + q];
    }
    __syncthreads();

    // out[n][j2] = pooled[n] . WoT[n][j2] + bo ; streams layout (bt, n, j2)
    for (int i = 0; i < 4; ++i) {
        int idx = tid + i * 256;
        int n = idx >> 8, j2 = idx & 255;
        const float* wr = WoT + ((size_t)(n * 256 + j2)) * 256;
        float acc = bo[n * 256 + j2];
        for (int k = 0; k < 256; ++k) acc += pooledSm[n * 256 + k] * wr[k];
        out[((size_t)bt * 4 + n) * 256 + j2] = acc;
    }
}

// =============== kernel 3: finalize diversity loss ===============
__global__ void k3_loss(const float* __restrict__ lossAcc, float* __restrict__ out)
{
    if (threadIdx.x == 0) out[131072] = 0.1f * (*lossAcc) / 768.f;  // /(6 pairs * 128 bt)
}

// =============== launch ===============
extern "C" void kernel_launch(void* const* d_in, const int* in_sizes, int n_in,
                              void* d_out, int out_size, void* d_ws, size_t ws_size,
                              hipStream_t stream)
{
    (void)in_sizes; (void)n_in; (void)out_size; (void)ws_size;
    const float* x     = (const float*)d_in[0];
    const float* g     = (const float*)d_in[1];
    const float* be    = (const float*)d_in[2];
    const float* query = (const float*)d_in[3];
    const float* Wk    = (const float*)d_in[4];
    const float* bk    = (const float*)d_in[5];
    const float* Wv    = (const float*)d_in[6];
    const float* bv    = (const float*)d_in[7];
    const float* Wo    = (const float*)d_in[8];
    const float* bo    = (const float*)d_in[9];
    float* out = (float*)d_out;

    char* ws = (char*)d_ws;
    f16*   G16     = (f16*)(ws + WS_G16);
    float* S       = (float*)(ws + WS_S);
    float* Cq      = (float*)(ws + WS_CQ);
    float* lossAcc = (float*)(ws + WS_LOSS);
    float* mu_ws   = (float*)(ws + WS_MU);
    float* rs_ws   = (float*)(ws + WS_RS);
    f16*   attn16  = (f16*)(ws + WS_ATTN16);
    f16*   Bv2     = (f16*)(ws + WS_BV2);
    float* WoT     = (float*)(ws + WS_WOT);

    hipLaunchKernelGGL(k0_prep, dim3(64), dim3(256), 0, stream,
                       g, be, query, Wk, bk, Wv, Wo, G16, S, Cq, Bv2, WoT, lossAcc);
    hipLaunchKernelGGL(k1_scores, dim3(BT_TOT), dim3(256), 0, stream,
                       x, G16, S, Cq, mu_ws, rs_ws, attn16, lossAcc);
    hipLaunchKernelGGL(k2_pool, dim3(BT_TOT), dim3(256), 0, stream,
                       x, g, be, mu_ws, rs_ws, attn16, Bv2, WoT, bv, bo, out);
    hipLaunchKernelGGL(k3_loss, dim3(1), dim3(32), 0, stream, lossAcc, out);
}
